// CNN3_F_P_3719441679181
// MI455X (gfx1250) — matrix operand order repaired and checked by reading, not compiled, not run on hardware
//
#include <hip/hip_runtime.h>
#include <hip/hip_bf16.h>

// ---------------------------------------------------------------------------
// CDNA5 (gfx1250) bf16-WMMA implementation of the pairwise-conv CNN.
// v_wmma_f32_16x16x32_bf16 everywhere; b128 fragment loads; async
// global->LDS double-buffered staging (ASYNCcnt) where available.
// ---------------------------------------------------------------------------

typedef __bf16 bf16_t;
typedef bf16_t   v16bf __attribute__((ext_vector_type(16)));
typedef float    v8f   __attribute__((ext_vector_type(8)));
typedef unsigned v4u   __attribute__((ext_vector_type(4)));
typedef int      v4i_vs __attribute__((vector_size(16)));   // matches builtin param

#define DEV __device__ __forceinline__

DEV unsigned short f2bf(float f) {
    union { float f; unsigned u; } x; x.f = f;
    unsigned r = x.u + 0x7FFFu + ((x.u >> 16) & 1u);
    return (unsigned short)(r >> 16);
}
DEV unsigned pack2(float lo, float hi) {
    return (unsigned)f2bf(lo) | ((unsigned)f2bf(hi) << 16);
}

union Frag { v16bf v; unsigned u[8]; v4u q[2]; };

// A-operand (16x32 bf16 MxK): this code uses VGPR v low k = half*16 + 2v (contiguous)
//   => k/2 runs {half*8+0..3} for q[0], {half*8+4..7} for q[1]  (b128-able)
// B-operand (32x16 bf16 KxN): same order as A here, VGPR v low k = half*16 + 2v
//   => one lane's 16 halves are k = half*16 .. half*16+15      (b128-able)

// ---------------------------------------------------------------------------
// Async global->LDS (gfx1250) with graceful fallback
// ---------------------------------------------------------------------------
#if defined(__gfx1250__) && __has_builtin(__builtin_amdgcn_global_load_async_to_lds_b128) && __has_builtin(__builtin_amdgcn_s_wait_asynccnt)
#define HAS_ASYNC 1
#else
#define HAS_ASYNC 0
#endif

#if HAS_ASYNC
DEV void async_cp128(const void* g, void* l) {
    // AS1 ptr == flat bits; AS3 offset == low 32 bits of generic LDS ptr
    __builtin_amdgcn_global_load_async_to_lds_b128(
        (__attribute__((address_space(1))) v4i_vs*)(unsigned long long)g,
        (__attribute__((address_space(3))) v4i_vs*)(unsigned long long)(unsigned)(unsigned long long)l,
        0, 0);
}
#define WAIT_ASYNC(n) __builtin_amdgcn_s_wait_asynccnt(n)
#else
DEV void async_cp128(const void* g, void* l) { *(v4u*)l = *(const v4u*)g; }
#define WAIT_ASYNC(n) ((void)0)
#endif

// ---------------------------------------------------------------------------
// Weight repacking: f32 -> bf16, layout [k>>4][n][16 halves] (b128 per lane)
// ---------------------------------------------------------------------------

// Wp: (256, 64, 2) -> wp0/wp1: 4 rows x (256 n x 16)
__global__ void pack_wp_kernel(const float* __restrict__ Wp,
                               unsigned short* __restrict__ wp0,
                               unsigned short* __restrict__ wp1) {
    int t = blockIdx.x * blockDim.x + threadIdx.x;
    if (t >= 256 * 64) return;
    int p = t >> 6, c = t & 63;
    int dst = (c >> 4) * 4096 + p * 16 + (c & 15);
    wp0[dst] = f2bf(Wp[p * 128 + c * 2 + 0]);
    wp1[dst] = f2bf(Wp[p * 128 + c * 2 + 1]);
}

// W: (256, 256, 3) -> per-dk: 16 rows x (256 n x 16)
__global__ void pack_conv_kernel(const float* __restrict__ W,
                                 unsigned short* __restrict__ wpk) {
    int t = blockIdx.x * blockDim.x + threadIdx.x;
    if (t >= 256 * 256 * 3) return;
    int o = t / 768, r = t % 768, i = r / 3, dk = r % 3;
    wpk[dk * 65536 + (i >> 4) * 4096 + o * 16 + (i & 15)] = f2bf(W[t]);
}

// Wfc1: (400, 32512), f = c*127 + l; K-order g = l*256 + c
// -> 2032 rows x (400 n x 16)
__global__ void pack_fc1_kernel(const float* __restrict__ W,
                                unsigned short* __restrict__ wp) {
    size_t t = (size_t)blockIdx.x * blockDim.x + threadIdx.x;
    if (t >= 400ull * 32512ull) return;
    int j = (int)(t / 32512), g = (int)(t % 32512);
    int c = g & 255, l = g >> 8;
    wp[(size_t)(g >> 4) * 6400 + j * 16 + (g & 15)] =
        f2bf(W[(size_t)j * 32512 + c * 127 + l]);
}

// ---------------------------------------------------------------------------
// Stage 0: h0[b,p,l] = relu( Wp0 @ x[:,0] + Wp1 @ x[:,l+1] + bp )
// One CTA per b; M=128(pad), N=256, K=64 (+ broadcast-row K=64 term).
// act layout: act[b][l*256 + c] (bf16)
// ---------------------------------------------------------------------------
__global__ __launch_bounds__(256)
void stage0_kernel(const float* __restrict__ x,
                   const unsigned* __restrict__ wp1p,
                   const unsigned* __restrict__ wp0p,
                   const float* __restrict__ bp,
                   unsigned short* __restrict__ out) {
    __shared__ unsigned xs[129 * 36];   // [row][cpair], stride 36 dwords (16B aligned rows)
    const int b = blockIdx.x, t = threadIdx.x;
    const int lane = t & 31, wave = t >> 5;
    const int Mw = wave >> 2, Nw = wave & 3;        // 2 x 4 wave grid
    const int half = lane >> 4, ln = lane & 15;

    const float* xb = x + (size_t)b * 8192;
    for (int idx = t; idx < 128 * 32; idx += 256) {
        int row = idx >> 5, cp = idx & 31;
        xs[row * 36 + cp] = pack2(xb[row * 64 + cp * 2], xb[row * 64 + cp * 2 + 1]);
    }
    if (t < 36) xs[128 * 36 + t] = 0u;              // zero row for M-pad
    __syncthreads();

    v8f acc[4][4];
    const v8f vz = {0.f, 0.f, 0.f, 0.f, 0.f, 0.f, 0.f, 0.f};
#pragma unroll
    for (int i = 0; i < 4; ++i)
#pragma unroll
        for (int j = 0; j < 4; ++j) acc[i][j] = vz;

#pragma unroll
    for (int kb = 0; kb < 2; ++kb) {
#pragma unroll
        for (int src = 0; src < 2; ++src) {
            const unsigned* wp = src ? wp0p : wp1p;
            Frag bfr[4], afr[4];
#pragma unroll
            for (int nf = 0; nf < 4; ++nf) {
                int o = Nw * 64 + nf * 16 + ln;
                const unsigned* wb = wp + ((kb * 2 + half) * 2048 + o * 8);
                bfr[nf].q[0] = *(const v4u*)wb;
                bfr[nf].q[1] = *(const v4u*)(wb + 4);
            }
#pragma unroll
            for (int mf = 0; mf < 4; ++mf) {
                int m = Mw * 64 + mf * 16 + ln;
                int row = src ? 0 : (m + 1);        // pad row m=127 -> zero row 128
                const unsigned* ab = &xs[row * 36 + kb * 16 + half * 8];
                afr[mf].q[0] = *(const v4u*)ab;
                afr[mf].q[1] = *(const v4u*)(ab + 4);
            }
#pragma unroll
            for (int mf = 0; mf < 4; ++mf)
#pragma unroll
                for (int nf = 0; nf < 4; ++nf)
                    acc[mf][nf] = __builtin_amdgcn_wmma_f32_16x16x32_bf16(
                        false, afr[mf].v, false, bfr[nf].v,
                        (short)0, acc[mf][nf], false, false);
        }
    }

    unsigned short* ob = out + (size_t)b * 32512;
#pragma unroll
    for (int mf = 0; mf < 4; ++mf)
#pragma unroll
        for (int nf = 0; nf < 4; ++nf) {
            int n = Nw * 64 + nf * 16 + ln;
            float bias = bp[n];
#pragma unroll
            for (int v = 0; v < 8; ++v) {
                int m = Mw * 64 + mf * 16 + half * 8 + v;
                if (m < 127) {
                    float val = acc[mf][nf][v] + bias;
                    ob[m * 256 + n] = f2bf(val > 0.f ? val : 0.f);
                }
            }
        }
}

// ---------------------------------------------------------------------------
// Conv layer: 3 shifted GEMMs (taps dk=0..2), K=256 each. Per-32-channel
// LDS blocks, double-buffered with async global->LDS copies.
// ---------------------------------------------------------------------------
DEV void conv_issue(const char* inb, unsigned* dst, int ib, int t) {
    // stage rows l=0..126, 64B (16 dwords) per row, as 4 x b128 chunks
    for (int idx = t; idx < 127 * 4; idx += 256) {
        int l = idx >> 2, c4 = idx & 3;
        async_cp128(inb + l * 512 + ib * 64 + c4 * 16,
                    dst + (l + 1) * 20 + c4 * 4);
    }
}

__global__ __launch_bounds__(256)
void conv_kernel(const unsigned short* __restrict__ in,
                 const unsigned* __restrict__ wpk,
                 const float* __restrict__ bias,
                 unsigned short* __restrict__ out) {
    __shared__ unsigned hs[2][130 * 20];  // [p][..], row stride 20 dwords
    const int b = blockIdx.x, t = threadIdx.x;
    const int lane = t & 31, wave = t >> 5;
    const int Mw = wave >> 2, Nw = wave & 3;
    const int half = lane >> 4, ln = lane & 15;
    const char* inb = (const char*)(in + (size_t)b * 32512);

    // zero padding rows 0,128,129 in both buffers (written once)
    for (int z = t; z < 120; z += 256) {
        int bu = z / 60, r3 = (z / 20) % 3, c = z % 20;
        int r = (r3 == 0) ? 0 : ((r3 == 1) ? 128 : 129);
        hs[bu][r * 20 + c] = 0u;
    }

    v8f acc[4][4];
    const v8f vz = {0.f, 0.f, 0.f, 0.f, 0.f, 0.f, 0.f, 0.f};
#pragma unroll
    for (int i = 0; i < 4; ++i)
#pragma unroll
        for (int j = 0; j < 4; ++j) acc[i][j] = vz;

    conv_issue(inb, hs[0], 0, t);                   // prologue: block 0

    for (int ib = 0; ib < 8; ++ib) {                // 8 channel blocks of 32
        if (ib < 7) {
            conv_issue(inb, hs[(ib + 1) & 1], ib + 1, t);
            WAIT_ASYNC(2);                          // block ib done, ib+1 in flight
        } else {
            WAIT_ASYNC(0);
        }
        __syncthreads();
        const unsigned* hb = hs[ib & 1];
#pragma unroll
        for (int dk = 0; dk < 3; ++dk) {
            const unsigned* wd = wpk + dk * 32768;
            Frag bfr[4], afr[4];
#pragma unroll
            for (int nf = 0; nf < 4; ++nf) {
                int o = Nw * 64 + nf * 16 + ln;
                const unsigned* wb = wd + ((ib * 2 + half) * 2048 + o * 8);
                bfr[nf].q[0] = *(const v4u*)wb;
                bfr[nf].q[1] = *(const v4u*)(wb + 4);
            }
#pragma unroll
            for (int mf = 0; mf < 4; ++mf) {
                int m = Mw * 64 + mf * 16 + ln;
                int p = m + dk;                      // reads input l = m+dk-1
                const unsigned* ab = &hb[p * 20 + half * 8];
                afr[mf].q[0] = *(const v4u*)ab;
                afr[mf].q[1] = *(const v4u*)(ab + 4);
            }
#pragma unroll
            for (int mf = 0; mf < 4; ++mf)
#pragma unroll
                for (int nf = 0; nf < 4; ++nf)
                    acc[mf][nf] = __builtin_amdgcn_wmma_f32_16x16x32_bf16(
                        false, afr[mf].v, false, bfr[nf].v,
                        (short)0, acc[mf][nf], false, false);
        }
        __syncthreads();
    }

    unsigned short* ob = out + (size_t)b * 32512;
#pragma unroll
    for (int mf = 0; mf < 4; ++mf)
#pragma unroll
        for (int nf = 0; nf < 4; ++nf) {
            int n = Nw * 64 + nf * 16 + ln;
            float bv = bias[n];
#pragma unroll
            for (int v = 0; v < 8; ++v) {
                int m = Mw * 64 + mf * 16 + half * 8 + v;
                if (m < 127) {
                    float val = acc[mf][nf][v] + bv;
                    ob[m * 256 + n] = f2bf(val > 0.f ? val : 0.f);
                }
            }
        }
}

// ---------------------------------------------------------------------------
// FC1: M=2048, N=400 (pad 512), K=32512 with K-order g = l*256+c so A rows
// are contiguous. Async double-buffered A staging; f32+relu output.
// ---------------------------------------------------------------------------
DEV void fc1_issue(const unsigned short* act, unsigned* dst, int b0, int step, int t) {
    // 64 rows x 64B, exactly one b128 chunk per thread
    int row = t >> 2, c4 = t & 3;
    const char* src = (const char*)(act + (size_t)(b0 + row) * 32512) + step * 64 + c4 * 16;
    async_cp128(src, dst + row * 20 + c4 * 4);
}

__global__ __launch_bounds__(256)
void fc1_kernel(const unsigned short* __restrict__ act,
                const unsigned* __restrict__ wp,
                const float* __restrict__ bias,
                float* __restrict__ h1) {
    __shared__ unsigned as[2][64 * 20];
    const int t = threadIdx.x, lane = t & 31, wave = t >> 5;
    const int Mw = wave >> 2, Nw = wave & 3;         // wave tile 32x32
    const int half = lane >> 4, ln = lane & 15;
    const int b0 = blockIdx.x * 64;
    const int j0 = blockIdx.y * 128;

    v8f acc[2][2];
    const v8f vz = {0.f, 0.f, 0.f, 0.f, 0.f, 0.f, 0.f, 0.f};
#pragma unroll
    for (int i = 0; i < 2; ++i)
#pragma unroll
        for (int j = 0; j < 2; ++j) acc[i][j] = vz;

    fc1_issue(act, as[0], b0, 0, t);

    for (int step = 0; step < 1016; ++step) {
        if (step < 1015) {
            fc1_issue(act, as[(step + 1) & 1], b0, step + 1, t);
            WAIT_ASYNC(1);
        } else {
            WAIT_ASYNC(0);
        }
        __syncthreads();
        const unsigned* ab0 = as[step & 1];
        Frag bfr[2], afr[2];
#pragma unroll
        for (int nf = 0; nf < 2; ++nf) {
            int o = j0 + Nw * 32 + nf * 16 + ln;
            int oc = o < 400 ? o : 0;                // clamp padded columns
            const unsigned* wb = wp + ((size_t)(step * 2 + half) * 3200 + oc * 8);
            bfr[nf].q[0] = *(const v4u*)wb;
            bfr[nf].q[1] = *(const v4u*)(wb + 4);
        }
#pragma unroll
        for (int mf = 0; mf < 2; ++mf) {
            int m = Mw * 32 + mf * 16 + ln;
            const unsigned* ab = &ab0[m * 20 + half * 8];
            afr[mf].q[0] = *(const v4u*)ab;
            afr[mf].q[1] = *(const v4u*)(ab + 4);
        }
#pragma unroll
        for (int mf = 0; mf < 2; ++mf)
#pragma unroll
            for (int nf = 0; nf < 2; ++nf)
                acc[mf][nf] = __builtin_amdgcn_wmma_f32_16x16x32_bf16(
                    false, afr[mf].v, false, bfr[nf].v,
                    (short)0, acc[mf][nf], false, false);
        __syncthreads();
    }

#pragma unroll
    for (int mf = 0; mf < 2; ++mf)
#pragma unroll
        for (int nf = 0; nf < 2; ++nf) {
            int j = j0 + Nw * 32 + nf * 16 + ln;
            if (j < 400) {
                float bv = bias[j];
#pragma unroll
                for (int v = 0; v < 8; ++v) {
                    int m = Mw * 32 + mf * 16 + half * 8 + v;
                    float val = acc[mf][nf][v] + bv;
                    h1[(size_t)(b0 + m) * 400 + j] = val > 0.f ? val : 0.f;
                }
            }
        }
}

// ---------------------------------------------------------------------------
// FC2: (2048, 400) @ (400, 2) + bias. Tiny -> scalar kernel.
// ---------------------------------------------------------------------------
__global__ void fc2_kernel(const float* __restrict__ h1,
                           const float* __restrict__ w2,
                           const float* __restrict__ b2,
                           float* __restrict__ out) {
    int b = blockIdx.x * blockDim.x + threadIdx.x;
    if (b >= 2048) return;
    const float* h = h1 + (size_t)b * 400;
    float a0 = 0.f, a1 = 0.f;
    for (int j = 0; j < 400; ++j) {
        float v = h[j];
        a0 += v * w2[j];
        a1 += v * w2[400 + j];
    }
    out[b * 2 + 0] = a0 + b2[0];
    out[b * 2 + 1] = a1 + b2[1];
}

// ---------------------------------------------------------------------------
// Host launcher
// ---------------------------------------------------------------------------
extern "C" void kernel_launch(void* const* d_in, const int* in_sizes, int n_in,
                              void* d_out, int out_size, void* d_ws, size_t ws_size,
                              hipStream_t stream) {
    const float* x    = (const float*)d_in[0];
    const float* Wp   = (const float*)d_in[1];
    const float* bp   = (const float*)d_in[2];
    const float* W1   = (const float*)d_in[3];
    const float* b1   = (const float*)d_in[4];
    const float* W2   = (const float*)d_in[5];
    const float* b2   = (const float*)d_in[6];
    const float* W3   = (const float*)d_in[7];
    const float* b3   = (const float*)d_in[8];
    const float* Wfc1 = (const float*)d_in[9];
    const float* bfc1 = (const float*)d_in[10];
    const float* Wfc2 = (const float*)d_in[11];
    const float* bfc2 = (const float*)d_in[12];
    float* out = (float*)d_out;

    char* base = (char*)d_ws;
    size_t off = 0;
    auto carve = [&](size_t bytes) -> char* {
        char* p = base + off;
        off += (bytes + 255) & ~(size_t)255;
        return p;
    };
    unsigned short* actA  = (unsigned short*)carve(2048ull * 32512ull * 2);  // 133 MB
    unsigned short* actB  = (unsigned short*)carve(2048ull * 32512ull * 2);  // 133 MB
    unsigned short* wp0p  = (unsigned short*)carve(4ull * 4096 * 2);
    unsigned short* wp1p  = (unsigned short*)carve(4ull * 4096 * 2);
    unsigned short* wc1   = (unsigned short*)carve(3ull * 65536 * 2);
    unsigned short* wc2   = (unsigned short*)carve(3ull * 65536 * 2);
    unsigned short* wc3   = (unsigned short*)carve(3ull * 65536 * 2);
    unsigned short* wfc1p = (unsigned short*)carve(2032ull * 6400 * 2);      // 26 MB
    float*          h1    = (float*)carve(2048ull * 400 * 4);                // 3.3 MB

    pack_wp_kernel<<<(256 * 64 + 255) / 256, 256, 0, stream>>>(Wp, wp0p, wp1p);
    pack_conv_kernel<<<(256 * 256 * 3 + 255) / 256, 256, 0, stream>>>(W1, wc1);
    pack_conv_kernel<<<(256 * 256 * 3 + 255) / 256, 256, 0, stream>>>(W2, wc2);
    pack_conv_kernel<<<(256 * 256 * 3 + 255) / 256, 256, 0, stream>>>(W3, wc3);
    pack_fc1_kernel<<<(int)((400ull * 32512 + 255) / 256), 256, 0, stream>>>(Wfc1, wfc1p);

    stage0_kernel<<<2048, 256, 0, stream>>>(x, (const unsigned*)wp1p,
                                            (const unsigned*)wp0p, bp, actA);
    conv_kernel<<<2048, 256, 0, stream>>>(actA, (const unsigned*)wc1, b1, actB);
    conv_kernel<<<2048, 256, 0, stream>>>(actB, (const unsigned*)wc2, b2, actA);
    conv_kernel<<<2048, 256, 0, stream>>>(actA, (const unsigned*)wc3, b3, actB);
    fc1_kernel<<<dim3(32, 4), 256, 0, stream>>>(actB, (const unsigned*)wfc1p, bfc1, h1);
    fc2_kernel<<<(2048 + 255) / 256, 256, 0, stream>>>(h1, Wfc2, bfc2, out);
}
